// AttentionSubBlock_76905684402462
// MI455X (gfx1250) — matrix operand order repaired and checked by reading, not compiled, not run on hardware
//
#include <hip/hip_runtime.h>

typedef __attribute__((ext_vector_type(16))) _Float16 v16h;
typedef __attribute__((ext_vector_type(8)))  float    v8f;

#define BB    8
#define TT    8
#define HH    14
#define WW    14
#define DIMC  768
#define HEADS 12
#define HD    64
#define NQ    1568           // 8*14*14
#define NKV   392            // 8*7*7
#define NKP   416            // padded to 13*32
#define MROWS (BB*NQ)        // 12544
#define EPSL  1e-6f
#define QSCALE 0.125f        // 64^-0.5

// ---------------- WMMA fragment helpers (16-bit, 16x16x32) ----------------
// A 16x32 f16: lane<16 row=lane, elems 0..7 = K0..7, 8..15 = K16..23
//              lane>=16 row=lane-16, elems = K8..15, K24..31
__device__ inline v16h load_a_frag(const _Float16* A, int ld, int row, int kk, int lane) {
  const _Float16* p = A + (size_t)row * ld + kk + ((lane >> 4) & 1) * 8;
  union { v16h v; uint4 u[2]; } f;
  f.u[0] = *(const uint4*)(p);
  f.u[1] = *(const uint4*)(p + 16);
  return f.v;
}
// B 32x16 f16 (stored as rows of length ld, row index = output column, contiguous K):
// lane col = lane%16, elems 0..7 = K kk+8*(lane>=16)+0..7, 8..15 = K kk+16+8*(lane>=16)+0..7
__device__ inline v16h load_b_frag(const _Float16* Wt, int ld, int col, int kk, int lane) {
  const _Float16* p = Wt + (size_t)col * ld + kk + ((lane >> 4) & 1) * 8;
  union { v16h v; uint4 u[2]; } f;
  f.u[0] = *(const uint4*)(p);
  f.u[1] = *(const uint4*)(p + 16);
  return f.v;
}

// ---------------- CDNA5 async global->LDS copy (ASYNCcnt path) ----------------
// Per ISA 10.x/15.18.3: per-lane LDS byte address in VDST VGPR, 64-bit global
// address in VADDR pair; 16 bytes per lane; tracked with ASYNCcnt.
__device__ inline void async_ld_b128(unsigned int lds_byte_addr, const void* gptr) {
  asm volatile("global_load_async_to_lds_b128 %0, %1, off"
               :
               : "v"(lds_byte_addr), "v"((unsigned long long)(uintptr_t)gptr)
               : "memory");
}
__device__ inline void wait_asynccnt0() {
  asm volatile("s_wait_asynccnt 0x0" ::: "memory");
}
// Flat LDS aperture: low 32 bits of generic pointer are the LDS byte address.
__device__ inline unsigned int lds_addr32(const void* p) {
  return (unsigned int)(uintptr_t)p;
}

// ---------------- elementwise f32 -> f16 ----------------
__global__ void cvt_f32_f16(const float* __restrict__ in, _Float16* __restrict__ out, int n) {
  int i = blockIdx.x * 256 + threadIdx.x;
  if (i < n) out[i] = (_Float16)in[i];
}

// ---------------- LayerNorm(DIM=768) -> f16, one wave per row ----------------
__global__ void __launch_bounds__(128) ln_to_f16(const float* __restrict__ x,
                                                 const float* __restrict__ g,
                                                 const float* __restrict__ b,
                                                 _Float16* __restrict__ out) {
  int wave = blockIdx.x * 4 + (threadIdx.x >> 5);
  int lane = threadIdx.x & 31;
  const float* xr = x + (size_t)wave * DIMC;
  float s = 0.f, ss = 0.f;
  for (int j = lane; j < DIMC; j += 32) { float v = xr[j]; s += v; ss += v * v; }
  for (int m = 16; m >= 1; m >>= 1) { s += __shfl_xor(s, m); ss += __shfl_xor(ss, m); }
  float mu = s / DIMC;
  float rs = rsqrtf(ss / DIMC - mu * mu + EPSL);
  _Float16* o = out + (size_t)wave * DIMC;
  for (int j = lane; j < DIMC; j += 32)
    o[j] = (_Float16)((xr[j] - mu) * rs * g[j] + b[j]);
}

// ---------------- WMMA GEMM: C[M,N] = A[M,K] * W[N,K]^T + bias ----------------
// block = 128 threads = 4 waves in 2x2; block tile 64x128; wave tile = 32x64.
// Per 32-K step the whole block stages A(64x32) + B(128x32) = 12KB into LDS via
// async global->LDS b128 copies. Chunk ownership is fixed per thread (2 A + 4 B
// chunks), so the issue sequence is branch-free and uniform; addresses advance
// by a constant 32-half stride per K-step.
__global__ void __launch_bounds__(128) gemm_f16_wmma(const _Float16* __restrict__ A,
                                                     const _Float16* __restrict__ W,
                                                     const float* __restrict__ bias,
                                                     float* __restrict__ C,
                                                     int M, int N, int K) {
  __shared__ __align__(16) _Float16 As[64 * 32];   // [row][k]  ld=32
  __shared__ __align__(16) _Float16 Bs[128 * 32];  // [col][k]  ld=32
  int tid   = threadIdx.x;
  int lane  = tid & 31;
  int wave  = tid >> 5;
  int blockM = blockIdx.y * 64;
  int blockN = blockIdx.x * 128;
  int waveM  = (wave >> 1) * 32;   // local within block tile
  int waveN  = (wave & 1) * 64;

  // fixed chunk slots: thread owns (row = tid>>2, quarter = tid&3) and +32-row/col copies
  int crow = tid >> 2, c4 = tid & 3;
  const _Float16* gA0 = A + (size_t)(blockM + crow) * K + c4 * 8;
  const _Float16* gA1 = gA0 + (size_t)32 * K;
  const _Float16* gB0 = W + (size_t)(blockN + crow) * K + c4 * 8;
  size_t bstep = (size_t)32 * K;
  unsigned lA0 = lds_addr32(&As[0]) + (unsigned)(crow * 32 + c4 * 8) * 2;
  unsigned lB0 = lds_addr32(&Bs[0]) + (unsigned)(crow * 32 + c4 * 8) * 2;

  v8f acc[2][4];
  v8f z = {};
  for (int mt = 0; mt < 2; ++mt)
    for (int nt = 0; nt < 4; ++nt) acc[mt][nt] = z;

  for (int kk = 0; kk < K; kk += 32) {
    // ---- stage tiles: 6 async b128 chunks per thread, no branches ----
    async_ld_b128(lA0,        gA0);
    async_ld_b128(lA0 + 2048, gA1);
    async_ld_b128(lB0,        gB0);
    async_ld_b128(lB0 + 2048, gB0 + bstep);
    async_ld_b128(lB0 + 4096, gB0 + 2 * bstep);
    async_ld_b128(lB0 + 6144, gB0 + 3 * bstep);
    gA0 += 32; gA1 += 32; gB0 += 32;
    wait_asynccnt0();
    __syncthreads();

    // ---- fragments from LDS ----
    v16h a0 = load_a_frag(As, 32, waveM + (lane & 15), 0, lane);
    v16h a1 = load_a_frag(As, 32, waveM + 16 + (lane & 15), 0, lane);
    v16h bf[4];
#pragma unroll
    for (int nt = 0; nt < 4; ++nt)
      bf[nt] = load_b_frag(Bs, 32, waveN + nt * 16 + (lane & 15), 0, lane);
#pragma unroll
    for (int nt = 0; nt < 4; ++nt) {
      acc[0][nt] = __builtin_amdgcn_wmma_f32_16x16x32_f16(false, a0, false, bf[nt],
                                                          (short)0, acc[0][nt], false, false);
      acc[1][nt] = __builtin_amdgcn_wmma_f32_16x16x32_f16(false, a1, false, bf[nt],
                                                          (short)0, acc[1][nt], false, false);
    }
    __syncthreads();  // protect LDS tiles before next iteration's async writes
  }

  for (int mt = 0; mt < 2; ++mt)
    for (int nt = 0; nt < 4; ++nt) {
      int n = blockN + waveN + nt * 16 + (lane & 15);
      float bv = bias[n];
      int rbase = blockM + waveM + mt * 16 + 8 * (lane >> 4);
#pragma unroll
      for (int r = 0; r < 8; ++r)
        C[(size_t)(rbase + r) * N + n] = acc[mt][nt][r] + bv;
    }
}

// ---------------- depthwise 3x3x3 conv pool + LayerNorm(HD), one wave/token ----------------
// mode 0: q  (dims 8,14,14, stride 1) -> q_pool f32 + q_f16 (scaled)
// mode 1: k  (dims 8,7,7, stride 1,2,2, padded to 416 rows) -> k_f16
// mode 2: v  (same dims) -> vT_f16 [64][416] transposed
__global__ void __launch_bounds__(128) pool_ln(const float* __restrict__ qkv,
                                               const float* __restrict__ cw,
                                               const float* __restrict__ ga,
                                               const float* __restrict__ be,
                                               int mode,
                                               float* __restrict__ outf32,
                                               _Float16* __restrict__ outf16) {
  int wave = blockIdx.x * 4 + (threadIdx.x >> 5);
  int lane = threadIdx.x & 31;
  int ntok = (mode == 0) ? NQ : NKP;
  int tok  = wave % ntok;
  int bh   = wave / ntok;
  int b    = bh / HEADS, head = bh % HEADS;
  int oh = (mode == 0) ? HH : 7, ow = (mode == 0) ? WW : 7;
  int sp = (mode == 0) ? 1 : 2;
  bool valid = (mode == 0) || (tok < NKV);
  float y0 = 0.f, y1 = 0.f;
  if (valid) {
    int t2 = tok / (oh * ow), h2 = (tok / ow) % oh, w2 = tok % ow;
    int coff = mode * DIMC + head * HD;
    int c0 = lane * 2;
#pragma unroll
    for (int cci = 0; cci < 2; ++cci) {
      int cc = c0 + cci;
      float acc = 0.f;
      for (int dt = 0; dt < 3; ++dt) {
        int ti = t2 + dt - 1;
        if (ti < 0 || ti >= TT) continue;
        for (int dh = 0; dh < 3; ++dh) {
          int hi = h2 * sp + dh - 1;
          if (hi < 0 || hi >= HH) continue;
          for (int dw = 0; dw < 3; ++dw) {
            int wi = w2 * sp + dw - 1;
            if (wi < 0 || wi >= WW) continue;
            acc += cw[cc * 27 + dt * 9 + dh * 3 + dw] *
                   qkv[((size_t)b * NQ + (ti * (HH * WW) + hi * WW + wi)) * (3 * DIMC) + coff + cc];
          }
        }
      }
      if (cci == 0) y0 = acc; else y1 = acc;
    }
    float s = y0 + y1, ss = y0 * y0 + y1 * y1;
    for (int m = 16; m >= 1; m >>= 1) { s += __shfl_xor(s, m); ss += __shfl_xor(ss, m); }
    float mu = s / HD;
    float rs = rsqrtf(ss / HD - mu * mu + EPSL);
    y0 = (y0 - mu) * rs * ga[lane * 2] + be[lane * 2];
    y1 = (y1 - mu) * rs * ga[lane * 2 + 1] + be[lane * 2 + 1];
  }
  if (mode == 0) {
    size_t base = ((size_t)bh * NQ + tok) * HD + lane * 2;
    outf32[base] = y0; outf32[base + 1] = y1;
    outf16[base] = (_Float16)(y0 * QSCALE); outf16[base + 1] = (_Float16)(y1 * QSCALE);
  } else if (mode == 1) {
    size_t base = ((size_t)bh * NKP + tok) * HD + lane * 2;
    outf16[base] = (_Float16)y0; outf16[base + 1] = (_Float16)y1;
  } else {
    size_t base = ((size_t)bh * HD + lane * 2) * NKP + tok;
    outf16[base] = (_Float16)y0; outf16[base + NKP] = (_Float16)y1;
  }
}

// ---------------- decomposed relative-position bias: one thread per (b,h,q,slot) ----------------
__global__ void rel_bias_kernel(const float* __restrict__ qp,
                                const float* __restrict__ rh,
                                const float* __restrict__ rw,
                                const float* __restrict__ rt,
                                float* __restrict__ bH, float* __restrict__ bW,
                                float* __restrict__ bT) {
  int tid = blockIdx.x * 256 + threadIdx.x;
  if (tid >= BB * HEADS * NQ * 22) return;
  int slot = tid % 22;
  int q    = (tid / 22) % NQ;
  int bh   = tid / (22 * NQ);
  int qt2 = q / (HH * WW), qh2 = (q / WW) % HH, qw2 = q % WW;
  const float* tab; int idx; float* outp;
  if (slot < 7)       { idx = qh2 - 2 * slot + 12;        tab = rh; outp = bH + ((size_t)bh * NQ + q) * 7 + slot; }
  else if (slot < 14) { int kw = slot - 7; idx = qw2 - 2 * kw + 12; tab = rw; outp = bW + ((size_t)bh * NQ + q) * 7 + kw; }
  else                { int kt = slot - 14; idx = qt2 - kt + 7;     tab = rt; outp = bT + ((size_t)bh * NQ + q) * 8 + kt; }
  const float* qv = qp + ((size_t)bh * NQ + q) * HD;
  const float* tv = tab + (size_t)idx * HD;
  float s = 0.f;
  for (int i = 0; i < HD; ++i) s += qv[i] * tv[i];
  *outp = s;
}

// ---------------- attention: 1 wave per 16 q rows of one (b,head) ----------------
__global__ void __launch_bounds__(32) attn_kernel(const _Float16* __restrict__ q_f16,
                                                  const _Float16* __restrict__ k_f16,
                                                  const _Float16* __restrict__ vT_f16,
                                                  const float* __restrict__ q_pool,
                                                  const float* __restrict__ bH,
                                                  const float* __restrict__ bW,
                                                  const float* __restrict__ bT,
                                                  _Float16* __restrict__ attn_out) {
  __shared__ __align__(16) float    S[16][NKP];
  __shared__ __align__(16) _Float16 P[16][NKP];
  __shared__ float bh_s[16][7], bw_s[16][7], bt_s[16][8];
  int lane = threadIdx.x;
  int qtile = blockIdx.x, head = blockIdx.y, b = blockIdx.z;
  int bhI = b * HEADS + head;
  int qbase = qtile * 16;
  const _Float16* qp = q_f16  + (size_t)bhI * NQ * HD;
  const _Float16* kp = k_f16  + (size_t)bhI * NKP * HD;
  const _Float16* vp = vT_f16 + (size_t)bhI * HD * NKP;

  // stage per-row decomposed biases
  for (int i = lane; i < 16 * 22; i += 32) {
    int r = i / 22, s = i % 22;
    size_t qi = (size_t)bhI * NQ + (qbase + r);
    if (s < 7)       bh_s[r][s]      = bH[qi * 7 + s];
    else if (s < 14) bw_s[r][s - 7]  = bW[qi * 7 + (s - 7)];
    else             bt_s[r][s - 14] = bT[qi * 8 + (s - 14)];
  }

  // scores S = (q*scale) @ k^T   (pads give 0, overwritten below)
  v16h a0 = load_a_frag(qp, HD, qbase + (lane & 15), 0, lane);
  v16h a1 = load_a_frag(qp, HD, qbase + (lane & 15), 32, lane);
  for (int kt = 0; kt < NKP / 16; ++kt) {
    v16h b0 = load_b_frag(kp, HD, kt * 16 + (lane & 15), 0, lane);
    v16h b1 = load_b_frag(kp, HD, kt * 16 + (lane & 15), 32, lane);
    v8f acc = {};
    acc = __builtin_amdgcn_wmma_f32_16x16x32_f16(false, a0, false, b0, (short)0, acc, false, false);
    acc = __builtin_amdgcn_wmma_f32_16x16x32_f16(false, a1, false, b1, (short)0, acc, false, false);
    int n = kt * 16 + (lane & 15);
#pragma unroll
    for (int r = 0; r < 8; ++r) S[r + 8 * (lane >> 4)][n] = acc[r];
  }
  __syncthreads();

  // bias add + mask + softmax; 2 lanes per row, 208 cols each
  int row = lane >> 1, hf = lane & 1;
  int cbeg = hf * 208, cend = cbeg + 208;
  float m = -1e30f;
  for (int c = cbeg; c < cend; ++c) {
    float s;
    if (c < NKV) {
      int kt3 = c / 49, rem = c % 49;
      s = S[row][c] + bh_s[row][rem / 7] + bw_s[row][rem % 7] + bt_s[row][kt3];
    } else s = -1e30f;
    S[row][c] = s;
    m = fmaxf(m, s);
  }
  m = fmaxf(m, __shfl_xor(m, 1));
  float sum = 0.f;
  for (int c = cbeg; c < cend; ++c) {
    float e = __expf(S[row][c] - m);
    S[row][c] = e;
    sum += e;
  }
  sum += __shfl_xor(sum, 1);
  float inv = 1.f / sum;
  for (int c = cbeg; c < cend; ++c) P[row][c] = (_Float16)(S[row][c] * inv);
  __syncthreads();

  // out = P @ V  (+ residual q_pool), write f16 [B*NQ][DIM]
  for (int nt = 0; nt < 4; ++nt) {
    v8f acc = {};
    for (int kk = 0; kk < NKP; kk += 32) {
      v16h pa = load_a_frag(&P[0][0], NKP, (lane & 15), kk, lane);
      v16h vb = load_b_frag(vp, NKP, nt * 16 + (lane & 15), kk, lane);
      acc = __builtin_amdgcn_wmma_f32_16x16x32_f16(false, pa, false, vb, (short)0, acc, false, false);
    }
    int n = nt * 16 + (lane & 15);
#pragma unroll
    for (int r = 0; r < 8; ++r) {
      int qr = qbase + r + 8 * (lane >> 4);
      float val = acc[r] + q_pool[((size_t)bhI * NQ + qr) * HD + n];
      attn_out[((size_t)(b * NQ + qr)) * DIMC + head * HD + n] = (_Float16)val;
    }
  }
}

// ---------------- orchestration ----------------
extern "C" void kernel_launch(void* const* d_in, const int* in_sizes, int n_in,
                              void* d_out, int out_size, void* d_ws, size_t ws_size,
                              hipStream_t stream) {
  (void)in_sizes; (void)n_in; (void)out_size; (void)ws_size;
  const float* x      = (const float*)d_in[0];
  const float* ln_g   = (const float*)d_in[1];
  const float* ln_b   = (const float*)d_in[2];
  const float* qkv_w  = (const float*)d_in[3];
  const float* qkv_b  = (const float*)d_in[4];
  const float* pq_w   = (const float*)d_in[5];
  const float* nq_g   = (const float*)d_in[6];
  const float* nq_b   = (const float*)d_in[7];
  const float* pk_w   = (const float*)d_in[8];
  const float* nk_g   = (const float*)d_in[9];
  const float* nk_b   = (const float*)d_in[10];
  const float* pv_w   = (const float*)d_in[11];
  const float* nv_g   = (const float*)d_in[12];
  const float* nv_b   = (const float*)d_in[13];
  const float* rel_h  = (const float*)d_in[14];
  const float* rel_w  = (const float*)d_in[15];
  const float* rel_t  = (const float*)d_in[16];
  const float* proj_w = (const float*)d_in[17];
  const float* proj_b = (const float*)d_in[18];
  float* out = (float*)d_out;

  char* w = (char*)d_ws;
  auto alloc = [&](size_t bytes) {
    void* p = (void*)w;
    w += (bytes + 255) & ~(size_t)255;
    return p;
  };
  _Float16* xn_f16   = (_Float16*)alloc((size_t)MROWS * DIMC * 2);
  _Float16* qkvw_f16 = (_Float16*)alloc((size_t)3 * DIMC * DIMC * 2);
  _Float16* projw_f16= (_Float16*)alloc((size_t)DIMC * DIMC * 2);
  float*    qkv_f32  = (float*)   alloc((size_t)MROWS * 3 * DIMC * 4);
  float*    q_pool   = (float*)   alloc((size_t)BB * HEADS * NQ * HD * 4);
  _Float16* q_f16    = (_Float16*)alloc((size_t)BB * HEADS * NQ * HD * 2);
  _Float16* k_f16    = (_Float16*)alloc((size_t)BB * HEADS * NKP * HD * 2);
  _Float16* vT_f16   = (_Float16*)alloc((size_t)BB * HEADS * HD * NKP * 2);
  float*    bH       = (float*)   alloc((size_t)BB * HEADS * NQ * 7 * 4);
  float*    bW       = (float*)   alloc((size_t)BB * HEADS * NQ * 7 * 4);
  float*    bT       = (float*)   alloc((size_t)BB * HEADS * NQ * 8 * 4);
  _Float16* attn_f16 = (_Float16*)alloc((size_t)MROWS * DIMC * 2);

  // 1. pre-LN -> f16
  ln_to_f16<<<MROWS / 4, 128, 0, stream>>>(x, ln_g, ln_b, xn_f16);
  // 2. weight converts
  {
    int n1 = 3 * DIMC * DIMC;
    cvt_f32_f16<<<(n1 + 255) / 256, 256, 0, stream>>>(qkv_w, qkvw_f16, n1);
    int n2 = DIMC * DIMC;
    cvt_f32_f16<<<(n2 + 255) / 256, 256, 0, stream>>>(proj_w, projw_f16, n2);
  }
  // 3. QKV GEMM: [12544 x 768] x [2304 x 768]^T  (async-LDS staged WMMA)
  gemm_f16_wmma<<<dim3((3 * DIMC) / 128, MROWS / 64), 128, 0, stream>>>(
      xn_f16, qkvw_f16, qkv_b, qkv_f32, MROWS, 3 * DIMC, DIMC);
  // 4. conv pools + LN
  pool_ln<<<(BB * HEADS * NQ) / 4, 128, 0, stream>>>(qkv_f32, pq_w, nq_g, nq_b, 0, q_pool, q_f16);
  pool_ln<<<(BB * HEADS * NKP) / 4, 128, 0, stream>>>(qkv_f32, pk_w, nk_g, nk_b, 1, nullptr, k_f16);
  pool_ln<<<(BB * HEADS * NKP) / 4, 128, 0, stream>>>(qkv_f32, pv_w, nv_g, nv_b, 2, nullptr, vT_f16);
  // 5. decomposed relative position biases
  {
    int tot = BB * HEADS * NQ * 22;
    rel_bias_kernel<<<(tot + 255) / 256, 256, 0, stream>>>(q_pool, rel_h, rel_w, rel_t, bH, bW, bT);
  }
  // 6. attention (softmax fp32 in LDS, WMMA for scores and PV)
  attn_kernel<<<dim3(NQ / 16, HEADS, BB), 32, 0, stream>>>(
      q_f16, k_f16, vT_f16, q_pool, bH, bW, bT, attn_f16);
  // 7. output projection -> d_out fp32
  gemm_f16_wmma<<<dim3(DIMC / 128, MROWS / 64), 128, 0, stream>>>(
      attn_f16, projw_f16, proj_b, out, MROWS, DIMC, DIMC);
}
